// SceneTracker_16922171146737
// MI455X (gfx1250) — compile-verified, matrix operand order checked by reading, not run on hardware
//
#include <hip/hip_runtime.h>
#include <hip/hip_bf16.h>

// ---------------- CDNA5 WMMA types ----------------
typedef __attribute__((ext_vector_type(16))) __bf16 v16bf;
typedef __attribute__((ext_vector_type(8)))  float  v8f;

#define S_    16
#define N_    1024
#define C_    128
#define H8_   48
#define W8_   64
#define NPIX  4080          // 3072 + 768 + 192 + 48
#define EDIM_ 460
#define OUTC  589           // 460 + 128 + 1

__device__ __forceinline__ unsigned short f2bf(float f) {
  unsigned int u = __float_as_uint(f);
  u += 0x7FFFu + ((u >> 16) & 1u);      // round-to-nearest-even
  return (unsigned short)(u >> 16);
}

union V16BF { v16bf v; uint4 q[2]; };

// A-matrix 16x32 bf16 per-lane load (ISA 7.12.2): lane&15 = row,
// (lane>>4)*8 = K sub-block; elements 0..7 -> K=kb..kb+7, 8..15 -> K=16+kb..
__device__ __forceinline__ v16bf load_bf_row(const unsigned short* row, int kk, int kbase) {
  V16BF r;
  r.q[0] = *(const uint4*)(row + kk * 32 + kbase);
  r.q[1] = *(const uint4*)(row + kk * 32 + 16 + kbase);
  return r.v;
}

// ---------------- K1: build bf16 pyramid + bf16 ffeats + W_ff^T ----------------
__global__ void k_prep(const float* __restrict__ fmaps, const float* __restrict__ ffeats,
                       const float* __restrict__ W_ff,
                       unsigned short* __restrict__ pyr, unsigned short* __restrict__ ff_bf,
                       unsigned short* __restrict__ wffT) {
  const int PYR_TOT = S_ * NPIX * C_;
  const int FF_TOT  = S_ * N_ * C_;
  int idx = blockIdx.x * blockDim.x + threadIdx.x;
  if (idx < PYR_TOT) {
    int c  = idx % C_;
    int ps = idx / C_;
    int p  = ps % NPIX;
    int s  = ps / NPIX;
    int lvl, lp;
    if      (p < 3072) { lvl = 0; lp = p; }
    else if (p < 3840) { lvl = 1; lp = p - 3072; }
    else if (p < 4032) { lvl = 2; lp = p - 3840; }
    else               { lvl = 3; lp = p - 4032; }
    int Wl = W8_ >> lvl;
    int hy = lp / Wl, wx = lp % Wl;
    int sz = 1 << lvl;
    const float* base = fmaps + (((size_t)(s * C_ + c) * H8_ + hy * sz) * W8_ + wx * sz);
    float acc = 0.f;
    for (int dy = 0; dy < sz; ++dy)
      for (int dx = 0; dx < sz; ++dx)
        acc += base[dy * W8_ + dx];
    pyr[idx] = f2bf(acc / (float)(sz * sz));
  } else if (idx < PYR_TOT + FF_TOT) {
    int i = idx - PYR_TOT;
    ff_bf[i] = f2bf(ffeats[i]);
  } else if (idx < PYR_TOT + FF_TOT + C_ * C_) {
    int i = idx - PYR_TOT - FF_TOT;
    int j = i / C_, c = i % C_;
    wffT[i] = f2bf(W_ff[c * C_ + j]);        // W_ff^T[j][c]
  }
}

// ---------------- K2: corr einsum for one s via WMMA bf16, 2x2 tile blocking ----
// corr[n, p] = (1/sqrt(C)) * sum_c ffeats[s,n,c] * pyr[s,p,c]
// Wave computes a 32(row) x 32(col) block (last col-pair is a 32x16 strip).
__global__ void k_corr(const unsigned short* __restrict__ ff_bf,
                       const unsigned short* __restrict__ pyr,
                       float* __restrict__ corr, int s) {
  int wave = threadIdx.x >> 5;
  int lane = threadIdx.x & 31;
  int tile = blockIdx.x * 4 + wave;          // [0, 4096): 32 m-pairs x 128 p-pairs
  int mp = tile & 31;
  int pp = tile >> 5;
  int n0 = mp * 32;
  int p0 = pp * 32;                          // pp==127 -> only 16 cols (p0=4064)
  bool has2 = (pp < 127);                    // wave-uniform
  int m = lane & 15;
  int kbase = (lane >> 4) * 8;
  const unsigned short* Arow0 = ff_bf + (size_t)(s * N_   + n0 + m) * C_;
  const unsigned short* Arow1 = Arow0 + 16 * C_;
  const unsigned short* Brow0 = pyr   + (size_t)(s * NPIX + p0 + m) * C_;
  const unsigned short* Brow1 = Brow0 + 16 * C_;
  __builtin_prefetch((const void*)(Brow0 + 64), 0, 0);   // global_prefetch_b8
  v8f acc00 = {}, acc01 = {}, acc10 = {}, acc11 = {};
#pragma unroll
  for (int kk = 0; kk < 4; ++kk) {
    v16bf a0 = load_bf_row(Arow0, kk, kbase);
    v16bf a1 = load_bf_row(Arow1, kk, kbase);
    v16bf b0 = load_bf_row(Brow0, kk, kbase);
    acc00 = __builtin_amdgcn_wmma_f32_16x16x32_bf16(false, a0, false, b0, (short)0, acc00,
                                                    false, false);
    acc10 = __builtin_amdgcn_wmma_f32_16x16x32_bf16(false, a1, false, b0, (short)0, acc10,
                                                    false, false);
    if (has2) {
      v16bf b1 = load_bf_row(Brow1, kk, kbase);
      acc01 = __builtin_amdgcn_wmma_f32_16x16x32_bf16(false, a0, false, b1, (short)0, acc01,
                                                      false, false);
      acc11 = __builtin_amdgcn_wmma_f32_16x16x32_bf16(false, a1, false, b1, (short)0, acc11,
                                                      false, false);
    }
  }
  const float scale = 0.08838834764831845f;  // 1/sqrt(128)
  int col = lane & 15;
  int rbase = 8 * (lane >> 4);
#pragma unroll
  for (int v = 0; v < 8; ++v) {
    size_t r0n = (size_t)(n0 + rbase + v);
    corr[r0n * NPIX + p0 + col]        = acc00[v] * scale;
    corr[(r0n + 16) * NPIX + p0 + col] = acc10[v] * scale;
    if (has2) {
      corr[r0n * NPIX + p0 + 16 + col]        = acc01[v] * scale;
      corr[(r0n + 16) * NPIX + p0 + 16 + col] = acc11[v] * scale;
    }
  }
}

// ---------------- K3: bilinear sample of corr -> fcorrs (raw) ----------------
__global__ void k_sample(const float* __restrict__ corr, const float* __restrict__ coords,
                         float* __restrict__ outp, int s) {
  int t = blockIdx.x * blockDim.x + threadIdx.x;
  if (t >= N_ * 196) return;
  int n = t / 196, q = t % 196;
  int lvl = q / 49, k = q % 49;
  float ddx = (float)(k % 7) - 3.f;
  float ddy = (float)(k / 7) - 3.f;
  float cx = coords[(size_t)(s * N_ + n) * 2 + 0];
  float cy = coords[(size_t)(s * N_ + n) * 2 + 1];
  float inv = 1.f / (float)(1 << lvl);
  float x = cx * inv + ddx;
  float y = cy * inv + ddy;
  int Wl = W8_ >> lvl, Hl = H8_ >> lvl;
  const int lvl_off[4] = {0, 3072, 3840, 4032};
  const float* cg = corr + (size_t)n * NPIX + lvl_off[lvl];
  float x0 = floorf(x), y0 = floorf(y);
  float fx = x - x0, fy = y - y0;
  int xi0 = (int)x0, yi0 = (int)y0;
  float w00 = (1.f - fx) * (1.f - fy);
  float w01 = (1.f - fx) * fy;
  float w10 = fx * (1.f - fy);
  float w11 = fx * fy;
  float acc = 0.f;
  if (xi0   >= 0 && xi0   < Wl && yi0   >= 0 && yi0   < Hl) acc += w00 * cg[yi0 * Wl + xi0];
  if (xi0   >= 0 && xi0   < Wl && yi0+1 >= 0 && yi0+1 < Hl) acc += w01 * cg[(yi0+1) * Wl + xi0];
  if (xi0+1 >= 0 && xi0+1 < Wl && yi0   >= 0 && yi0   < Hl) acc += w10 * cg[yi0 * Wl + xi0+1];
  if (xi0+1 >= 0 && xi0+1 < Wl && yi0+1 >= 0 && yi0+1 < Hl) acc += w11 * cg[(yi0+1) * Wl + xi0+1];
  outp[((size_t)n * S_ + s) * OUTC + 130 + lvl * 49 + k] = acc;
}

// ---------------- K4: embeddings, dep flows, dcorrs, ffeats copy, concat;
//                  add pos_e + times_e to all of x[0:460] ----------------
__global__ void k_misc(const float* __restrict__ coords, const float* __restrict__ deps_pred,
                       const float* __restrict__ fdeps, const float* __restrict__ ffeats,
                       const float* __restrict__ track_mask, const float* __restrict__ vis_init,
                       float* __restrict__ outp) {
  int bid = blockIdx.x;
  int n = bid >> 4, s = bid & 15;
  __shared__ float sc[6];
  if (threadIdx.x == 0) {
    float cx  = coords[(size_t)(s * N_ + n) * 2 + 0];
    float cy  = coords[(size_t)(s * N_ + n) * 2 + 1];
    float cx0 = coords[(size_t)n * 2 + 0];
    float cy0 = coords[(size_t)n * 2 + 1];
    float dep  = fmaxf(deps_pred[s * N_ + n], 0.01f);
    float dep0 = fmaxf(deps_pred[n], 0.01f);
    // invs_proj: bilinear sample of 1/fdeps (fdeps<0.01 -> 85) at (cx,cy), level 0
    float x0 = floorf(cx), y0 = floorf(cy);
    float fx = cx - x0, fy = cy - y0;
    int xi0 = (int)x0, yi0 = (int)y0;
    float wgt[4] = {(1.f-fx)*(1.f-fy), (1.f-fx)*fy, fx*(1.f-fy), fx*fy};
    int xs[4] = {xi0, xi0, xi0+1, xi0+1};
    int ys[4] = {yi0, yi0+1, yi0, yi0+1};
    const float* fd = fdeps + (size_t)s * H8_ * W8_;
    float ip = 0.f;
    for (int ci = 0; ci < 4; ++ci) {
      int xi = xs[ci], yi = ys[ci];
      if (xi >= 0 && xi < W8_ && yi >= 0 && yi < H8_) {
        float f = fd[yi * W8_ + xi];
        float fin = (f < 0.01f) ? (1.0f / 85.0f) : (1.0f / f);
        ip += wgt[ci] * fin;
      }
    }
    sc[0] = cx - cx0;          // dx
    sc[1] = cy - cy0;          // dy
    sc[2] = dep - dep0;        // dep_flow
    sc[3] = ip - 1.f / dep;    // dcorr
    sc[4] = cx0; sc[5] = cy0;
  }
  __syncthreads();
  float dx = sc[0], dy = sc[1], depflow = sc[2], dcorr = sc[3];
  float cx0 = sc[4], cy0 = sc[5];
  size_t o = ((size_t)n * S_ + s) * OUTC;
  const float LN1E4 = 9.210340371976184f;
  for (int e = threadIdx.x; e < EDIM_; e += blockDim.x) {
    // ---- pos_e[n][e]: bilinear of 2D sincos grid at (cx0, cy0) ----
    float pe;
    {
      int e2 = e; bool use_x = (e2 < 230); if (!use_x) e2 -= 230;
      bool is_sin = (e2 < 115); int j = is_sin ? e2 : e2 - 115;
      float om = __expf(-(float)j * (LN1E4 / 115.f));
      float x0 = floorf(cx0), y0 = floorf(cy0);
      float fx = cx0 - x0, fy = cy0 - y0;
      int xi0 = (int)x0, yi0 = (int)y0;
      float wgt[4] = {(1.f-fx)*(1.f-fy), (1.f-fx)*fy, fx*(1.f-fy), fx*fy};
      int xs[4] = {xi0, xi0, xi0+1, xi0+1};
      int ys[4] = {yi0, yi0+1, yi0, yi0+1};
      float acc = 0.f;
      for (int ci = 0; ci < 4; ++ci) {
        int xi = xs[ci], yi = ys[ci];
        if (xi >= 0 && xi < W8_ && yi >= 0 && yi < H8_) {
          float ang = (use_x ? (float)xi : (float)yi) * om;
          acc += wgt[ci] * (is_sin ? sinf(ang) : cosf(ang));
        }
      }
      pe = acc;
      // ---- times_e[s][e] ----
      int jt = (e < 230) ? e : e - 230;
      float omt = __expf(-(float)jt * (LN1E4 / 230.f));
      float angt = (float)s * omt;
      pe += (e < 230) ? sinf(angt) : cosf(angt);
    }
    // ---- ti element ----
    float base;
    if (e == 0)       base = dx;
    else if (e == 1)  base = dy;
    else if (e < 66)  { int i = e - 2;  int j = i >> 1; float a = dx * (31.25f * (float)j);
                        base = (i & 1) ? cosf(a) : sinf(a); }
    else if (e < 130) { int i = e - 66; int j = i >> 1; float a = dy * (31.25f * (float)j);
                        base = (i & 1) ? cosf(a) : sinf(a); }
    else if (e < 326) base = outp[o + e];                 // fcorrs from k_sample
    else if (e < 328) base = depflow;
    else if (e < 330) base = dcorr;
    else if (e < 458) base = ffeats[(size_t)(s * N_ + n) * C_ + (e - 330)];
    else {
      // faithful reproduction of the reference's (B,S,2N,1)->(B*N,S,2) reshape
      int c = e - 458;
      int src_s, m;
      if (s < 8) { src_s = 2 * s + c;      m = 2 * n;     }
      else       { src_s = 2 * s + c - 16; m = 2 * n + 1; }
      base = (m < N_) ? track_mask[src_s * N_ + m] : vis_init[src_s * N_ + (m - N_)];
    }
    outp[o + e] = base + pe;
  }
}

// ---------------- K5: GroupNorm(LayerNorm over C) -> bf16 gn ----------------
__global__ void k_gn(const float* __restrict__ delta_feats, const float* __restrict__ gamma,
                     const float* __restrict__ beta, unsigned short* __restrict__ gn_bf) {
  int gtid = blockIdx.x * blockDim.x + threadIdx.x;
  int r = gtid >> 5;                 // row = n*16 + s
  int lane = threadIdx.x & 31;
  if (r >= N_ * S_) return;
  int n = r >> 4, s = r & 15;
  const float* src = delta_feats + (size_t)(s * N_ + n) * C_;
  float v[4]; float sum = 0.f, sq = 0.f;
#pragma unroll
  for (int j = 0; j < 4; ++j) {
    v[j] = src[lane + 32 * j];
    sum += v[j]; sq += v[j] * v[j];
  }
#pragma unroll
  for (int off = 16; off; off >>= 1) {
    sum += __shfl_xor(sum, off, 32);
    sq  += __shfl_xor(sq,  off, 32);
  }
  float mu  = sum * (1.f / 128.f);
  float var = sq * (1.f / 128.f) - mu * mu;
  float inv = rsqrtf(var + 1e-5f);
#pragma unroll
  for (int j = 0; j < 4; ++j) {
    int c = lane + 32 * j;
    float g = (v[j] - mu) * inv * gamma[c] + beta[c];
    gn_bf[(size_t)r * C_ + c] = f2bf(g);
  }
}

// ---------------- K6: gn @ W_ff via WMMA (wave = 16x128 strip),
//                  GELU(exact) + residual ----------------
__global__ void k_gemm2(const unsigned short* __restrict__ gn_bf,
                        const unsigned short* __restrict__ wffT,
                        const float* __restrict__ b_ff, const float* __restrict__ ffeats,
                        float* __restrict__ outp) {
  int wave = threadIdx.x >> 5;
  int lane = threadIdx.x & 31;
  int rtile = blockIdx.x * 4 + wave;  // [0, 1024)
  int r0 = rtile * 16;
  int m = lane & 15;
  int kbase = (lane >> 4) * 8;
  const unsigned short* Arow = gn_bf + (size_t)(r0 + m) * C_;
  v8f acc[8] = {};
#pragma unroll
  for (int kk = 0; kk < 4; ++kk) {
    v16bf a = load_bf_row(Arow, kk, kbase);
#pragma unroll
    for (int jt = 0; jt < 8; ++jt) {
      v16bf b = load_bf_row(wffT + (size_t)(jt * 16 + m) * C_, kk, kbase);
      acc[jt] = __builtin_amdgcn_wmma_f32_16x16x32_bf16(false, a, false, b, (short)0,
                                                        acc[jt], false, false);
    }
  }
  int rbase = 8 * (lane >> 4);
#pragma unroll
  for (int jt = 0; jt < 8; ++jt) {
    int j = jt * 16 + (lane & 15);
    float bias = b_ff[j];
#pragma unroll
    for (int v = 0; v < 8; ++v) {
      int r = r0 + rbase + v;
      float t = acc[jt][v] + bias;
      float g = 0.5f * t * (1.0f + erff(t * 0.7071067811865475f));   // exact GELU
      int n = r >> 4, s = r & 15;
      float ffn = g + ffeats[(size_t)(s * N_ + n) * C_ + j];
      outp[(size_t)r * OUTC + 460 + j] = ffn;
    }
  }
}

// ---------------- K7: vis head (ffeats_new @ W_vis + b_vis) ----------------
__global__ void k_vis(const float* __restrict__ W_vis, const float* __restrict__ b_vis,
                      float* __restrict__ outp) {
  int gtid = blockIdx.x * blockDim.x + threadIdx.x;
  int r = gtid >> 5;
  int lane = threadIdx.x & 31;
  if (r >= N_ * S_) return;
  const float* row = outp + (size_t)r * OUTC + 460;
  float sum = 0.f;
#pragma unroll
  for (int j = 0; j < 4; ++j) {
    int c = lane + 32 * j;
    sum += row[c] * W_vis[c];
  }
#pragma unroll
  for (int off = 16; off; off >>= 1) sum += __shfl_xor(sum, off, 32);
  if (lane == 0) outp[(size_t)r * OUTC + 588] = sum + b_vis[0];
}

// ---------------- launch ----------------
extern "C" void kernel_launch(void* const* d_in, const int* in_sizes, int n_in,
                              void* d_out, int out_size, void* d_ws, size_t ws_size,
                              hipStream_t stream) {
  const float* fmaps      = (const float*)d_in[0];
  const float* fdeps      = (const float*)d_in[1];
  const float* coords     = (const float*)d_in[2];
  const float* deps_pred  = (const float*)d_in[3];
  const float* ffeats     = (const float*)d_in[4];
  const float* delta_f    = (const float*)d_in[5];
  const float* track_mask = (const float*)d_in[6];
  const float* vis_init   = (const float*)d_in[7];
  const float* gn_gamma   = (const float*)d_in[8];
  const float* gn_beta    = (const float*)d_in[9];
  const float* W_ff       = (const float*)d_in[10];
  const float* b_ff       = (const float*)d_in[11];
  const float* W_vis      = (const float*)d_in[12];
  const float* b_vis      = (const float*)d_in[13];
  float* outp = (float*)d_out;

  // workspace carve-up (256B-aligned sub-buffers), total ~42 MB
  char* ws = (char*)d_ws;
  const size_t PYR_B  = (size_t)S_ * NPIX * C_ * 2;   // 16,711,680
  const size_t FF_B   = (size_t)S_ * N_ * C_ * 2;     //  4,194,304
  const size_t WFT_B  = (size_t)C_ * C_ * 2;          //     32,768
  const size_t GN_B   = (size_t)N_ * S_ * C_ * 2;     //  4,194,304
  unsigned short* pyr   = (unsigned short*)(ws);
  unsigned short* ff_bf = (unsigned short*)(ws + PYR_B);
  unsigned short* wffT  = (unsigned short*)(ws + PYR_B + FF_B);
  unsigned short* gn_bf = (unsigned short*)(ws + PYR_B + FF_B + WFT_B);
  float*          corr  = (float*)         (ws + PYR_B + FF_B + WFT_B + GN_B);

  // K1: prep bf16 pyramid / ffeats / W_ff^T
  {
    int total = S_ * NPIX * C_ + S_ * N_ * C_ + C_ * C_;
    int blocks = (total + 255) / 256;
    k_prep<<<blocks, 256, 0, stream>>>(fmaps, ffeats, W_ff, pyr, ff_bf, wffT);
  }
  // per-s: WMMA corr einsum (2x2 blocked), then bilinear sampling (scratch reused)
  for (int s = 0; s < S_; ++s) {
    k_corr<<<1024, 128, 0, stream>>>(ff_bf, pyr, corr, s);
    k_sample<<<(N_ * 196 + 255) / 256, 256, 0, stream>>>(corr, coords, outp, s);
  }
  // K4: everything else in x[0:460] (+pos/time embeddings added everywhere)
  k_misc<<<N_ * S_, 128, 0, stream>>>(coords, deps_pred, fdeps, ffeats,
                                      track_mask, vis_init, outp);
  // K5-K7: GN -> WMMA GEMM + GELU + residual -> vis head
  k_gn<<<(N_ * S_ * 32 + 255) / 256, 256, 0, stream>>>(delta_f, gn_gamma, gn_beta, gn_bf);
  k_gemm2<<<256, 128, 0, stream>>>(gn_bf, wffT, b_ff, ffeats, outp);
  k_vis<<<(N_ * S_ * 32 + 255) / 256, 256, 0, stream>>>(W_vis, b_vis, outp);

  (void)in_sizes; (void)n_in; (void)out_size; (void)ws_size;
}
